// APPNP10Net_3375844295350
// MI455X (gfx1250) — matrix operand order repaired and checked by reading, not compiled, not run on hardware
//
#include <hip/hip_runtime.h>
#include <hip/hip_fp16.h>

typedef __attribute__((ext_vector_type(16))) _Float16 v16h;
typedef __attribute__((ext_vector_type(8)))  float    v8f;
typedef __attribute__((ext_vector_type(4)))  float    v4f;

#define F_IN  512
#define HID   64
#define CDIM  40
#define ALPHAF 0.1f
#define KSTEPS 10

// ---------------------------------------------------------------------------
// Pre-swizzle w1 (f32 [512][64] row-major) into a contiguous-k WMMA B-operand
// layout for V_WMMA_F32_16X16X32_F16, as f16. For k-chunk kc (32 rows of K),
// n-tile t (16 cols), lane L, the 16 halves a lane needs are stored
// contiguously so the GEMM kernel does one 32B load per B tile.
//   element e (0..15) of lane L = w1[kc*32 + (L>>4)*16 + e][t*16 + (L&15)]
// ---------------------------------------------------------------------------
__global__ void k_w1_swizzle(const float* __restrict__ w1,
                             _Float16* __restrict__ w1s) {
    int idx = blockIdx.x * blockDim.x + threadIdx.x;   // 0 .. F_IN*HID-1
    if (idx >= F_IN * HID) return;
    int e  = idx & 15;
    int L  = (idx >> 4) & 31;
    int t  = (idx >> 9) & 3;
    int kc = idx >> 11;
    int k = kc * 32 + ((L >> 4) * 16) + e;
    int n = t * 16 + (L & 15);
    w1s[idx] = (_Float16)w1[k * HID + n];
}

// ---------------------------------------------------------------------------
// Fused MLP: h0 = relu(x @ w1 + b1) @ w2 + b2     (also copied into hA)
// One wave per 16-row tile; WMMA 16x16x32 f16->f32 for GEMM1 (K=512, N=64),
// relu'd tile staged in LDS, scalar GEMM2 (K=64, N=40) from LDS.
// ---------------------------------------------------------------------------
__global__ void __launch_bounds__(256)
k_mlp_wmma(const float* __restrict__ x, const _Float16* __restrict__ w1s,
           const float* __restrict__ b1, const float* __restrict__ w2,
           const float* __restrict__ b2, float* __restrict__ h0,
           float* __restrict__ hA, int N) {
    __shared__ float lds_h[8][16][68];      // padded: bank-conflict-free
    __shared__ float lds_w2[HID * CDIM];
    __shared__ float lds_b2[CDIM];

    const int tid  = threadIdx.x;
    const int lane = tid & 31;
    const int wave = tid >> 5;

    for (int i = tid; i < HID * CDIM; i += 256) lds_w2[i] = w2[i];
    if (tid < CDIM) lds_b2[tid] = b2[tid];
    __syncthreads();

    const int rowBase = (blockIdx.x * 8 + wave) * 16;
    const int m    = lane & 15;          // A-matrix row for this lane
    const int koff = (lane >> 4) * 16;   // lanes 16-31 hold K+16..31
    int rowA = rowBase + m;
    if (rowA > N - 1) rowA = N - 1;      // clamp (WMMA needs EXEC all-ones)
    const float* xrow = x + (size_t)rowA * F_IN;

    v8f acc[4] = {};                     // four 16x16 f32 accumulators (N=64)

    #pragma unroll 4
    for (int kc = 0; kc < F_IN / 32; ++kc) {
        // A tile: lane needs x[m][kc*32+koff .. +15]
        const float* ap = xrow + kc * 32 + koff;
        v4f g0 = *(const v4f*)(ap);
        v4f g1 = *(const v4f*)(ap + 4);
        v4f g2 = *(const v4f*)(ap + 8);
        v4f g3 = *(const v4f*)(ap + 12);
        v16h a;
        #pragma unroll
        for (int j = 0; j < 4; ++j) {
            a[j]      = (_Float16)g0[j];
            a[4 + j]  = (_Float16)g1[j];
            a[8 + j]  = (_Float16)g2[j];
            a[12 + j] = (_Float16)g3[j];
        }
        #pragma unroll
        for (int t = 0; t < 4; ++t) {
            v16h b = *(const v16h*)(w1s + ((size_t)(kc * 4 + t) * 32 + lane) * 16);
            acc[t] = __builtin_amdgcn_wmma_f32_16x16x32_f16(
                false, a, false, b, (short)0, acc[t], false, false);
        }
    }

    // bias + relu, spill 16x64 tile to LDS.
    // C/D layout: acc[t][r] = C[r + 8*(lane>>4)][t*16 + (lane&15)]
    const int hi = lane >> 4;
    #pragma unroll
    for (int t = 0; t < 4; ++t) {
        const int col  = t * 16 + (lane & 15);
        const float bv = b1[col];
        #pragma unroll
        for (int r = 0; r < 8; ++r) {
            float v = acc[t][r] + bv;
            v = v > 0.0f ? v : 0.0f;
            lds_h[wave][r + 8 * hi][col] = v;
        }
    }
    __syncthreads();

    // GEMM2: 16x40 outputs per wave, 64-deep dot products from LDS
    for (int o = lane; o < 16 * CDIM; o += 32) {
        const int r = o / CDIM, c = o % CDIM;
        float sum = lds_b2[c];
        #pragma unroll 8
        for (int k = 0; k < HID; ++k)
            sum += lds_h[wave][r][k] * lds_w2[k * CDIM + c];
        const int grow = rowBase + r;
        if (grow < N) {
            const size_t oi = (size_t)grow * CDIM + c;
            h0[oi] = sum;
            hA[oi] = sum;
        }
    }
}

// ---------------------------------------------------------------------------
// Degree / normalization kernels.  dinv buffer doubles as deg accumulator.
// deg includes the self-loop, so init to 1.0 then atomic-add per dst edge.
// ---------------------------------------------------------------------------
__global__ void k_fill1(float* __restrict__ d, int N) {
    int i = blockIdx.x * blockDim.x + threadIdx.x;
    if (i < N) d[i] = 1.0f;
}

__global__ void k_deg(const int* __restrict__ dst, float* __restrict__ deg, int E) {
    int e = blockIdx.x * blockDim.x + threadIdx.x;
    if (e < E) atomicAdd(&deg[dst[e]], 1.0f);
}

__global__ void k_rsq(float* __restrict__ d, int N) {
    int i = blockIdx.x * blockDim.x + threadIdx.x;
    if (i < N) d[i] = rsqrtf(d[i]);          // deg >= 1 always (self-loop)
}

__global__ void k_norm(const int* __restrict__ src, const int* __restrict__ dst,
                       const float* __restrict__ dinv, float* __restrict__ nrm, int E) {
    int e = blockIdx.x * blockDim.x + threadIdx.x;
    if (e < E) nrm[e] = dinv[src[e]] * dinv[dst[e]];
}

// h_next = (1-a)*(self-loop term) + a*h0 ; scatter kernel adds edge terms
__global__ void k_step_init(const float* __restrict__ hc, const float* __restrict__ h0,
                            const float* __restrict__ dinv, float* __restrict__ hn, int N) {
    int t = blockIdx.x * blockDim.x + threadIdx.x;
    if (t >= N * CDIM) return;
    int row = t / CDIM;
    float d = dinv[row];
    hn[t] = (1.0f - ALPHAF) * (hc[t] * d * d) + ALPHAF * h0[t];
}

// one thread per (edge, channel); 40 consecutive threads share an edge so the
// gather of hc[src] is a coalesced 160B segment (L2-resident working set)
__global__ void k_scatter(const int* __restrict__ src, const int* __restrict__ dst,
                          const float* __restrict__ nrm, const float* __restrict__ hc,
                          float* __restrict__ hn, int E) {
    long long t = (long long)blockIdx.x * blockDim.x + threadIdx.x;
    if (t >= (long long)E * CDIM) return;
    int e = (int)(t / CDIM);
    int c = (int)(t % CDIM);
    float v = (1.0f - ALPHAF) * nrm[e] * hc[(size_t)src[e] * CDIM + c];
    atomicAdd(&hn[(size_t)dst[e] * CDIM + c], v);
}

__global__ void k_logsoftmax(const float* __restrict__ h, float* __restrict__ out, int N) {
    int row = blockIdx.x * blockDim.x + threadIdx.x;
    if (row >= N) return;
    const float* hr = h + (size_t)row * CDIM;
    float mx = hr[0];
    for (int c = 1; c < CDIM; ++c) mx = fmaxf(mx, hr[c]);
    float s = 0.0f;
    for (int c = 0; c < CDIM; ++c) s += __expf(hr[c] - mx);
    float lse = mx + __logf(s);
    for (int c = 0; c < CDIM; ++c) out[(size_t)row * CDIM + c] = hr[c] - lse;
}

// ---------------------------------------------------------------------------
extern "C" void kernel_launch(void* const* d_in, const int* in_sizes, int n_in,
                              void* d_out, int out_size, void* d_ws, size_t ws_size,
                              hipStream_t stream) {
    const float* x     = (const float*)d_in[0];
    const int*   edges = (const int*)d_in[1];
    const float* w1    = (const float*)d_in[2];
    const float* b1    = (const float*)d_in[3];
    const float* w2    = (const float*)d_in[4];
    const float* b2    = (const float*)d_in[5];
    float*       out   = (float*)d_out;

    const int N = in_sizes[0] / F_IN;
    const int E = in_sizes[1] / 2;
    const int* srcI = edges;
    const int* dstI = edges + E;

    // workspace carve (256B aligned sections)
    char* p = (char*)d_ws;
    auto align_up = [](size_t v) { return (v + 255) & ~(size_t)255; };
    _Float16* w1s = (_Float16*)p; p += align_up((size_t)F_IN * HID * sizeof(_Float16));
    float* h0   = (float*)p;      p += align_up((size_t)N * CDIM * sizeof(float));
    float* hA   = (float*)p;      p += align_up((size_t)N * CDIM * sizeof(float));
    float* hB   = (float*)p;      p += align_up((size_t)N * CDIM * sizeof(float));
    float* dinv = (float*)p;      p += align_up((size_t)N * sizeof(float));
    float* nrm  = (float*)p;      p += align_up((size_t)E * sizeof(float));
    (void)ws_size; (void)n_in; (void)out_size;

    // 1. weight swizzle + fused MLP (WMMA)
    k_w1_swizzle<<<(F_IN * HID + 255) / 256, 256, 0, stream>>>(w1, w1s);
    k_mlp_wmma<<<(N + 127) / 128, 256, 0, stream>>>(x, w1s, b1, w2, b2, h0, hA, N);

    // 2. GCN normalization
    k_fill1<<<(N + 255) / 256, 256, 0, stream>>>(dinv, N);
    k_deg<<<(E + 255) / 256, 256, 0, stream>>>(dstI, dinv, E);
    k_rsq<<<(N + 255) / 256, 256, 0, stream>>>(dinv, N);
    k_norm<<<(E + 255) / 256, 256, 0, stream>>>(srcI, dstI, dinv, nrm, E);

    // 3. K propagation steps (ping-pong hA/hB)
    float* hc = hA;
    float* hn = hB;
    const int iBlocks = (N * CDIM + 255) / 256;
    const long long totS = (long long)E * CDIM;
    const int sBlocks = (int)((totS + 255) / 256);
    for (int s = 0; s < KSTEPS; ++s) {
        k_step_init<<<iBlocks, 256, 0, stream>>>(hc, h0, dinv, hn, N);
        k_scatter<<<sBlocks, 256, 0, stream>>>(srcI, dstI, nrm, hc, hn, E);
        float* tmp = hc; hc = hn; hn = tmp;
    }

    // 4. log-softmax
    k_logsoftmax<<<(N + 255) / 256, 256, 0, stream>>>(hc, out, N);
}
